// Transformer_549755814623
// MI455X (gfx1250) — matrix operand order repaired and checked by reading, not compiled, not run on hardware
//
#include <hip/hip_runtime.h>
#include <hip/hip_bf16.h>
#include <math.h>

typedef __attribute__((ext_vector_type(16))) _Float16 v16h;
typedef __attribute__((ext_vector_type(8)))  float    v8f;
typedef __attribute__((ext_vector_type(4)))  unsigned int v4u;

// ---------------- problem constants ----------------
constexpr int kB   = 128;   // batch
constexpr int kLs  = 168;   // source length
constexpr int kTp  = 24;    // target length
constexpr int kD   = 64;    // input dim
constexpr int kC   = 128;   // TCN channels
constexpr int kW   = 169;   // window length Ls+1
constexpr int kL   = 168;   // kv length
constexpr int kN   = kTp * kB;  // 3072 sequences

// LDS activation geometry (time-major: Act[w][c], fp16)
constexpr int kPAD   = 8;              // left zero pad rows (max causal shift = 2*4)
constexpr int kWPAD  = 176;            // padded column count (11 tiles of 16)
constexpr int kROWS  = kPAD + kWPAD;   // 184
constexpr int kSX    = 64;             // X0 row stride (halves)
constexpr int kSA    = 128;            // A/B row stride (halves)

// fp16 weight tiles in workspace: 16x32 A-fragments, lane-major (lane*16+i), 1024 halves/tile
constexpr int OFF_C01 = 0;           // tcn0_w1: 3 taps * 8 mt * 2 kt = 48 tiles
constexpr int OFF_C02 = 48  * 1024;  // tcn0_w2: 3*8*4 = 96
constexpr int OFF_C0D = 144 * 1024;  // tcn0_wd: 1*8*2 = 16
constexpr int OFF_C11 = 160 * 1024;  // 96
constexpr int OFF_C12 = 256 * 1024;  // 96
constexpr int OFF_C21 = 352 * 1024;  // 96
constexpr int OFF_C22 = 448 * 1024;  // 96
constexpr int kTILES  = 544;

// dynamic LDS layout
constexpr int kHALVES    = kROWS * (kSX + kSA + kSA);       // 58880 halves
constexpr int kFLOAT_OFF = kHALVES * 2;                      // 117760 bytes
constexpr int kNFLOATS   = 176*6 + 128 + 256*2 + 8;          // sc,xv,xs,cs,cs2,tau,cvec,red,red2,spare
constexpr int kSMEM      = kFLOAT_OFF + kNFLOATS * 4;        // ~124.6 KB

// ---------------- weight prep: f32 -> fp16 A-fragment layout ----------------
__global__ void prep_weights(const float* __restrict__ w01, const float* __restrict__ w02,
                             const float* __restrict__ w0d, const float* __restrict__ w11,
                             const float* __restrict__ w12, const float* __restrict__ w21,
                             const float* __restrict__ w22, _Float16* __restrict__ ws) {
  int gid = blockIdx.x * blockDim.x + threadIdx.x;
  if (gid >= kTILES * 1024) return;
  int tileId = gid >> 10;
  int e    = gid & 1023;
  int lane = e >> 4;
  int i    = e & 15;
  // A-matrix 16x32 f16 element mapping (wave32): m = lane&15; g = lane>>4
  int m = lane & 15, g = lane >> 4;
  int v = i >> 1, h = i & 1;
  int k = g * 16 + v * 2 + h;

  const float* w; int base, KT, Cin, K;
  if      (tileId < 48)  { w = w01; base = 0;   KT = 2; Cin = 64;  K = 3; }
  else if (tileId < 144) { w = w02; base = 48;  KT = 4; Cin = 128; K = 3; }
  else if (tileId < 160) { w = w0d; base = 144; KT = 2; Cin = 64;  K = 1; }
  else if (tileId < 256) { w = w11; base = 160; KT = 4; Cin = 128; K = 3; }
  else if (tileId < 352) { w = w12; base = 256; KT = 4; Cin = 128; K = 3; }
  else if (tileId < 448) { w = w21; base = 352; KT = 4; Cin = 128; K = 3; }
  else                   { w = w22; base = 448; KT = 4; Cin = 128; K = 3; }
  int local = tileId - base;
  int kt  = local % KT;
  int mt  = (local / KT) & 7;
  int tap = local / (KT * 8);
  int c   = mt * 16 + m;
  int cin = kt * 32 + k;
  ws[gid] = (_Float16)w[((size_t)c * Cin + cin) * K + tap];
}

// ---------------- fused conv pass (WMMA) ----------------
// Y[c,w] = relu(bias[c] + sum_tap sum_cin W_tap[c,cin] * In[cin, w - (TAPS-1-tap)*dil] [+ residual])
// RES: 0 = none, 1 = 1x1 GEMM residual from resL (stride kSX, 2 ktiles) with bres, 2 = identity add from resL (stride kSA)
template <int TAPS, int KT, int RES>
__device__ __forceinline__ void conv_pass(const _Float16* __restrict__ inL, int inStride,
                                          _Float16* __restrict__ outL,
                                          const _Float16* __restrict__ wt_, const float* __restrict__ bias,
                                          int dil,
                                          const _Float16* __restrict__ resL,
                                          const _Float16* __restrict__ wres,
                                          const float* __restrict__ bres, int tid) {
  const int lane = tid & 31;
  const int mt   = tid >> 5;       // wave id = output-channel tile (8 waves, 8 tiles)
  const int nI   = lane & 15;      // output column within tile
  const int g    = lane >> 4;

  constexpr int NMAIN = TAPS * KT;
  constexpr int NB    = NMAIN + ((RES == 1) ? 2 : 0);

  // hoist weight A-fragments for this channel tile (reused across 11 column tiles)
  v16h afr[NMAIN];
#pragma unroll
  for (int tap = 0; tap < TAPS; ++tap)
#pragma unroll
    for (int kt = 0; kt < KT; ++kt)
      afr[tap * KT + kt] = *(const v16h*)(wt_ + (size_t)((tap * 8 + mt) * KT + kt) * 1024 + lane * 16);

  v16h rfr[2];
  if (RES == 1) {
#pragma unroll
    for (int kt = 0; kt < 2; ++kt)
      rfr[kt] = *(const v16h*)(wres + (size_t)(mt * 2 + kt) * 1024 + lane * 16);
  }

  float bval[8];
#pragma unroll
  for (int r = 0; r < 8; ++r) {
    int ch = mt * 16 + g * 8 + r;
    bval[r] = bias[ch] + ((RES == 1) ? bres[ch] : 0.0f);
  }

  for (int w0 = 0; w0 < kWPAD; w0 += 16) {
    // --- load ALL B-fragments for this tile first (SSA-distinct -> batched ds_loads,
    //     WMMA chain then drains with partial dscnt waits instead of wait-0 per op) ---
    v16h bfr[NB];
#pragma unroll
    for (int tap = 0; tap < TAPS; ++tap) {
      const int shift = (TAPS - 1 - tap) * dil;
      const _Float16* brow = inL + (kPAD + w0 + nI - shift) * inStride + g * 16;
#pragma unroll
      for (int kt = 0; kt < KT; ++kt)
        bfr[tap * KT + kt] = *(const v16h*)(brow + kt * 32);
    }
    if (RES == 1) {
      const _Float16* brow = resL + (kPAD + w0 + nI) * kSX + g * 16;
#pragma unroll
      for (int kt = 0; kt < 2; ++kt)
        bfr[NMAIN + kt] = *(const v16h*)(brow + kt * 32);
    }

    v8f acc;
#pragma unroll
    for (int r = 0; r < 8; ++r) acc[r] = bval[r];

#pragma unroll
    for (int i = 0; i < NMAIN; ++i)
      acc = __builtin_amdgcn_wmma_f32_16x16x32_f16(false, afr[i], false, bfr[i],
                                                   (short)0, acc, false, false);
    if (RES == 1) {
#pragma unroll
      for (int kt = 0; kt < 2; ++kt)
        acc = __builtin_amdgcn_wmma_f32_16x16x32_f16(false, rfr[kt], false, bfr[NMAIN + kt],
                                                     (short)0, acc, false, false);
    }

    const int orow = (kPAD + w0 + nI) * kSA + mt * 16 + g * 8;
    if (RES == 2) {  // identity residual (in-place on A is safe: per-lane read==write address)
#pragma unroll
      for (int r = 0; r < 8; ++r) acc[r] += (float)resL[orow + r];
    }
    union { _Float16 h[8]; v4u u; } pk;
#pragma unroll
    for (int r = 0; r < 8; ++r) pk.h[r] = (_Float16)fmaxf(acc[r], 0.0f);
    *(v4u*)(outL + orow) = pk.u;  // ds_store_b128
  }
}

// ---------------- fused per-sequence kernel ----------------
__global__ void __launch_bounds__(256)
fused_tcn_attn(const float* __restrict__ src, const float* __restrict__ trg,
               const _Float16* __restrict__ ws,
               const float* __restrict__ b01, const float* __restrict__ b02, const float* __restrict__ b0d,
               const float* __restrict__ b11, const float* __restrict__ b12,
               const float* __restrict__ b21, const float* __restrict__ b22,
               const float* __restrict__ h1w, const float* __restrict__ h1b,
               const float* __restrict__ h2w, const float* __restrict__ h2b,
               float* __restrict__ out) {
  extern __shared__ __align__(32) char smem[];
  _Float16* X0 = (_Float16*)smem;            // [kROWS][64]
  _Float16* A  = X0 + kROWS * kSX;           // [kROWS][128]
  _Float16* Bf = A + kROWS * kSA;            // [kROWS][128]
  float* sc   = (float*)(smem + kFLOAT_OFF);
  float* xv   = sc + 176;
  float* xs   = xv + 176;
  float* cs   = xs + 176;
  float* cs2  = cs + 176;
  float* tau  = cs2 + 176;
  float* cvec = tau + 176;
  float* red  = cvec + 128;
  float* red2 = red + 256;

  const int tid = threadIdx.x;
  const int n   = blockIdx.x;   // sequence id = t*B + b
  const int t   = n >> 7;
  const int b   = n & 127;

  // zero all half buffers (incl. causal pad rows + column tail)
  {
    v4u z; z.x = z.y = z.z = z.w = 0u;
    v4u* p = (v4u*)smem;
    for (int i = tid; i < kHALVES / 8; i += 256) p[i] = z;
  }
  __syncthreads();

  // gather input window: X0[w][d] = srctrg[b, t+w, d], cast to fp16
  for (int idx = tid; idx < kW * kD; idx += 256) {
    int w = idx >> 6, d = idx & 63;
    int tt = t + w;
    float v = (tt < kLs) ? src[((size_t)b * kLs + tt) * kD + d]
                         : trg[((size_t)b * kTp + (tt - kLs)) * kD + d];
    X0[(kPAD + w) * kSX + d] = (_Float16)v;
  }
  __syncthreads();

  // TCN block 0 (dil 1, with 1x1 downsample residual)
  conv_pass<3, 2, 0>(X0, kSX, Bf, ws + OFF_C01, b01, 1, nullptr, nullptr, nullptr, tid);
  __syncthreads();
  conv_pass<3, 4, 1>(Bf, kSA, A, ws + OFF_C02, b02, 1, X0, ws + OFF_C0D, b0d, tid);
  __syncthreads();
  // TCN block 1 (dil 2, identity residual)
  conv_pass<3, 4, 0>(A, kSA, Bf, ws + OFF_C11, b11, 2, nullptr, nullptr, nullptr, tid);
  __syncthreads();
  conv_pass<3, 4, 2>(Bf, kSA, A, ws + OFF_C12, b12, 2, A, nullptr, nullptr, tid);
  __syncthreads();
  // TCN block 2 (dil 4, identity residual)
  conv_pass<3, 4, 0>(A, kSA, Bf, ws + OFF_C21, b21, 4, nullptr, nullptr, nullptr, tid);
  __syncthreads();
  conv_pass<3, 4, 2>(Bf, kSA, A, ws + OFF_C22, b22, 4, A, nullptr, nullptr, tid);
  __syncthreads();

  // ---- attention scores: s[l] = q . kv[l],  q = row 168, kv = rows 0..167 ----
  if (tid < kL) {
    const _Float16* kvr = A + (kPAD + tid) * kSA;
    const _Float16* qr  = A + (kPAD + kL) * kSA;
    float s = 0.0f;
#pragma unroll 4
    for (int c = 0; c < kC; ++c) s += (float)kvr[c] * (float)qr[c];
    sc[tid] = s;
  }
  __syncthreads();

  // ---- entmax 1.5 over 168 values ----
  red[tid] = (tid < kL) ? sc[tid] : -3.4e38f;
  __syncthreads();
  for (int s = 128; s > 0; s >>= 1) {
    if (tid < s) red[tid] = fmaxf(red[tid], red[tid + s]);
    __syncthreads();
  }
  float mx = red[0];
  __syncthreads();
  if (tid < kL) xv[tid] = (sc[tid] - mx) * 0.5f;
  __syncthreads();
  // stable descending rank sort
  if (tid < kL) {
    float xi = xv[tid];
    int rank = 0;
    for (int j = 0; j < kL; ++j) {
      float xj = xv[j];
      rank += (xj > xi) || (xj == xi && j < tid);
    }
    xs[rank] = xi;
  }
  __syncthreads();
  if (tid < kL) { cs[tid] = xs[tid]; cs2[tid] = xs[tid] * xs[tid]; }
  __syncthreads();
  for (int off = 1; off < kL; off <<= 1) {  // Hillis-Steele inclusive scans
    float a = 0.0f, a2 = 0.0f;
    if (tid < kL && tid >= off) { a = cs[tid - off]; a2 = cs2[tid - off]; }
    __syncthreads();
    if (tid < kL) { cs[tid] += a; cs2[tid] += a2; }
    __syncthreads();
  }
  float flag = 0.0f;
  if (tid < kL) {
    float rho  = (float)(tid + 1);
    float mean = cs[tid] / rho;
    float msq  = cs2[tid] / rho;
    float ssv  = rho * (msq - mean * mean);
    float del  = (1.0f - ssv) / rho;
    float ta   = mean - sqrtf(fmaxf(del, 0.0f));
    tau[tid] = ta;
    flag = (ta <= xs[tid]) ? 1.0f : 0.0f;
  }
  red[tid] = flag;
  __syncthreads();
  for (int s = 128; s > 0; s >>= 1) {
    if (tid < s) red[tid] += red[tid + s];
    __syncthreads();
  }
  int support = (int)(red[0] + 0.5f);
  if (support < 1) support = 1;
  __syncthreads();
  float tstar = tau[support - 1];
  if (tid < kL) {
    float d = fmaxf(xv[tid] - tstar, 0.0f);
    sc[tid] = d * d;  // entmax probabilities
  }
  __syncthreads();

  // ---- context vector c[c] = sum_l score[l]*kv[l,c] ----
  if (tid < kC) {
    float acc = 0.0f;
    for (int l = 0; l < kL; ++l) acc += sc[l] * (float)A[(kPAD + l) * kSA + tid];
    cvec[tid] = acc;
  }
  __syncthreads();

  // ---- heads on cc = [c, q] (256) ----
  float ccv = (tid < 128) ? cvec[tid] : (float)A[(kPAD + kL) * kSA + (tid - 128)];
  red[tid]  = h1w[tid] * ccv;
  red2[tid] = h2w[tid] * ccv;
  __syncthreads();
  for (int s = 128; s > 0; s >>= 1) {
    if (tid < s) { red[tid] += red[tid + s]; red2[tid] += red2[tid + s]; }
    __syncthreads();
  }
  if (tid == 0) {
    float y = red[0] + h1b[0];
    float z = red2[0] + h2b[0];
    float sp = (z > 20.0f) ? z : log1pf(expf(z));  // softplus
    out[(size_t)b * kTp + t] = y;                          // y_hat (B,Tp)
    out[(size_t)kB * kTp + (size_t)b * kTp + t] = sp;      // sigma (B,Tp)
  }
}

// ---------------- host launch ----------------
extern "C" void kernel_launch(void* const* d_in, const int* in_sizes, int n_in,
                              void* d_out, int out_size, void* d_ws, size_t ws_size,
                              hipStream_t stream) {
  (void)in_sizes; (void)n_in; (void)out_size; (void)ws_size;
  const float* src = (const float*)d_in[0];
  const float* trg = (const float*)d_in[1];
  const float* w01 = (const float*)d_in[2];
  const float* b01 = (const float*)d_in[3];
  const float* w02 = (const float*)d_in[4];
  const float* b02 = (const float*)d_in[5];
  const float* w0d = (const float*)d_in[6];
  const float* b0d = (const float*)d_in[7];
  const float* w11 = (const float*)d_in[8];
  const float* b11 = (const float*)d_in[9];
  const float* w12 = (const float*)d_in[10];
  const float* b12 = (const float*)d_in[11];
  const float* w21 = (const float*)d_in[12];
  const float* b21 = (const float*)d_in[13];
  const float* w22 = (const float*)d_in[14];
  const float* b22 = (const float*)d_in[15];
  const float* h1w = (const float*)d_in[16];
  const float* h1b = (const float*)d_in[17];
  const float* h2w = (const float*)d_in[18];
  const float* h2b = (const float*)d_in[19];
  _Float16* ws = (_Float16*)d_ws;
  float* out = (float*)d_out;

  (void)hipFuncSetAttribute((const void*)fused_tcn_attn,
                            hipFuncAttributeMaxDynamicSharedMemorySize, kSMEM);

  // 1) convert/relayout weights into WMMA A-fragment order (fp16) in workspace
  {
    int total = kTILES * 1024;
    prep_weights<<<(total + 255) / 256, 256, 0, stream>>>(w01, w02, w0d, w11, w12, w21, w22, ws);
  }
  // 2) one workgroup per sequence: fused TCN (WMMA) + entmax attention + heads
  fused_tcn_attn<<<dim3(kN), dim3(256), kSMEM, stream>>>(
      src, trg, ws, b01, b02, b0d, b11, b12, b21, b22, h1w, h1b, h2w, h2b, out);
}
